// Tree_82162724372765
// MI455X (gfx1250) — matrix operand order repaired and checked by reading, not compiled, not run on hardware
//
#include <hip/hip_runtime.h>
#include <stdint.h>

typedef __attribute__((ext_vector_type(16))) _Float16 v16h;
typedef __attribute__((ext_vector_type(8)))  float    v8f;
typedef __attribute__((ext_vector_type(8)))  uint32_t v8u;

#define S_DIM 1024
#define E_DIM 1024
#define A_DIM 16
#define H_DIM 4096
#define M_SIM 16
#define D_MAX 16
#define KB1   2065
#define KB1P  2176   /* padded to a multiple of 128 for the x4-unrolled K loop */

#define ACT_NONE 0
#define ACT_TANH 1
#define ACT_RELU 2

// Scalar slots (int view of ws scalar block)
// 0=node 1=done 2=l_node 3=l_act 4=plen 5=action 6=bk_nd 7=bk_valid 8=rew(float)

// -------- pack W[K][N] f32 row-major -> Wp[(Kpad/2)][N], each u32 = half2{W[k],W[k+1]} --------
__global__ void pack_f16_kernel(const float* __restrict__ W, uint32_t* __restrict__ Wp,
                                int K, int Kpad, int N) {
  size_t total = (size_t)(Kpad >> 1) * (size_t)N;
  size_t stride = (size_t)gridDim.x * blockDim.x;
  for (size_t idx = (size_t)blockIdx.x * blockDim.x + threadIdx.x; idx < total; idx += stride) {
    size_t r = idx / (size_t)N;
    size_t n = idx - r * (size_t)N;
    int k = (int)(r << 1);
    float lo = (k < K) ? W[(size_t)k * N + n] : 0.0f;
    float hi = (k + 1 < K) ? W[(size_t)(k + 1) * N + n] : 0.0f;
    union { _Float16 h[2]; uint32_t u; } cv;
    cv.h[0] = (_Float16)lo;
    cv.h[1] = (_Float16)hi;
    Wp[idx] = cv.u;
  }
}

// -------- GEMV via v_wmma_f32_16x16x32_f16: y[n] = act(bias[n] + sum_k x[k]*W[k][n]) --------
// One wave handles 32 output columns (two 16-wide tiles sharing one A fragment).
// x lives in matrix row M=0 of the A fragment. K loop unrolled x4 (Kpad % 128 == 0).
__global__ void __launch_bounds__(32)
gemv_wmma_kernel(const uint32_t* __restrict__ Wp, long long wstride, const int* __restrict__ widx,
                 const float* __restrict__ xb, int xstride, const int* __restrict__ xidx,
                 const float* __restrict__ bias, int bstride, const int* __restrict__ bidx,
                 float* __restrict__ y, int Kpad, int N, int act) {
  if (widx) Wp += (long long)(*widx) * wstride;
  if (xidx) xb += (size_t)(*xidx) * (size_t)xstride;
  if (bidx) bias += (size_t)(*bidx) * (size_t)bstride;
  const int lane = threadIdx.x;   // wave32
  const int m    = lane & 15;     // tile column / A-matrix M id
  const int kh   = lane >> 4;     // K-half selector (contiguous k = 16*kh + e, same in A and B)
  const int n0   = blockIdx.x * 32;
  const bool two = (n0 + 16) < N; // wave-uniform: second tile exists (EXEC stays all-ones)
  v8f acc0 = {}, acc1 = {};
  for (int k0 = 0; k0 < Kpad; k0 += 128) {
#pragma unroll
    for (int u = 0; u < 4; ++u) {
      const int kb = k0 + u * 32;
      // A fragment (16x32 f16): vector x only in row M=0.
      // lanes<16 carry K = kb+{0..15}; lanes>=16 carry K = kb+{16..31}.
      v16h a;
#pragma unroll
      for (int e = 0; e < 16; ++e) {
        int kk = kb + kh * 16 + e;
        float xv = xb[kk];                     // uniform per half-wave -> broadcast
        a[e] = (m == 0) ? (_Float16)xv : (_Float16)0.0f;
      }
      // B fragments (32x16 f16): lanes 0-15 hold K 0-15, lanes 16-31 hold K 16-31,
      // packed 2 K-values per VGPR -> coalesced u32 loads from packed weights.
      const uint32_t* wp = Wp + ((size_t)(kb >> 1) + (size_t)kh * 8) * (size_t)N + (n0 + m);
      __builtin_prefetch(wp + (size_t)16 * N, 0, 1);   // next K-tile -> global_prefetch_b8
      v8u braw0;
#pragma unroll
      for (int j = 0; j < 8; ++j) braw0[j] = wp[(size_t)j * N];
      v16h b0 = __builtin_bit_cast(v16h, braw0);
      acc0 = __builtin_amdgcn_wmma_f32_16x16x32_f16(false, a, false, b0, (short)0, acc0,
                                                    false, false);
      if (two) {
        v8u braw1;
#pragma unroll
        for (int j = 0; j < 8; ++j) braw1[j] = wp[(size_t)j * N + 16];
        v16h b1 = __builtin_bit_cast(v16h, braw1);
        acc1 = __builtin_amdgcn_wmma_f32_16x16x32_f16(false, a, false, b1, (short)0, acc1,
                                                      false, false);
      }
    }
  }
  if (lane < 16) {                      // C/D row M=0 lives in acc[0] of lanes 0-15
    float r = acc0[0];
    if (bias) r += bias[n0 + lane];
    if (act == ACT_TANH) r = tanhf(r);
    else if (act == ACT_RELU) r = fmaxf(r, 0.0f);
    y[n0 + lane] = r;
    if (two) {
      float r1 = acc1[0];
      if (bias) r1 += bias[n0 + 16 + lane];
      if (act == ACT_TANH) r1 = tanhf(r1);
      else if (act == ACT_RELU) r1 = fmaxf(r1, 0.0f);
      y[n0 + 16 + lane] = r1;
    }
  }
}

// -------- threefry2x32 (JAX PRNG) --------
__device__ __forceinline__ uint32_t rotl32(uint32_t x, int r) { return (x << r) | (x >> (32 - r)); }

__device__ void threefry2x32(uint32_t k0, uint32_t k1, uint32_t x0, uint32_t x1,
                             uint32_t* o0, uint32_t* o1) {
  uint32_t ks0 = k0, ks1 = k1, ks2 = k0 ^ k1 ^ 0x1BD11BDAu;
  const int ra[4] = {13, 15, 26, 6};
  const int rb[4] = {17, 29, 16, 24};
  x0 += ks0; x1 += ks1;
#pragma unroll
  for (int i = 0; i < 4; ++i) { x0 += x1; x1 = rotl32(x1, ra[i]); x1 ^= x0; }
  x0 += ks1; x1 += ks2 + 1u;
#pragma unroll
  for (int i = 0; i < 4; ++i) { x0 += x1; x1 = rotl32(x1, rb[i]); x1 ^= x0; }
  x0 += ks2; x1 += ks0 + 2u;
#pragma unroll
  for (int i = 0; i < 4; ++i) { x0 += x1; x1 = rotl32(x1, ra[i]); x1 ^= x0; }
  x0 += ks0; x1 += ks1 + 3u;
#pragma unroll
  for (int i = 0; i < 4; ++i) { x0 += x1; x1 = rotl32(x1, rb[i]); x1 ^= x0; }
  x0 += ks1; x1 += ks2 + 4u;
#pragma unroll
  for (int i = 0; i < 4; ++i) { x0 += x1; x1 = rotl32(x1, ra[i]); x1 ^= x0; }
  x0 += ks2; x1 += ks0 + 5u;
  *o0 = x0; *o1 = x1;
}

// -------- categorical sample (Gumbel argmax, JAX semantics) + reward dot --------
__global__ void trav_sample_kernel(int* __restrict__ sc, const float* __restrict__ logits,
                                   const float* __restrict__ states,
                                   const float* __restrict__ w_rew, int m, int d) {
  __shared__ float red[256];
  __shared__ int s_act;
  if (threadIdx.x == 0) {
    uint32_t k0, k1, kd0, kd1;
    threefry2x32(0u, 42u, 0u, (uint32_t)m, &k0, &k1);   // kk = fold_in(key(42), m)
    threefry2x32(k0, k1, 0u, (uint32_t)d, &kd0, &kd1);  // fold_in(kk, d)
    uint32_t bits[16];
#pragma unroll
    for (int i = 0; i < 8; ++i) {
      uint32_t a, b;
      threefry2x32(kd0, kd1, (uint32_t)i, (uint32_t)(i + 8), &a, &b);
      bits[i] = a; bits[i + 8] = b;
    }
    const float TINY = 1.1754943508222875e-38f;
    float best = -__builtin_inff(); int bi = 0;
    for (int i = 0; i < A_DIM; ++i) {
      union { uint32_t u; float f; } cv;
      cv.u = (bits[i] >> 9) | 0x3f800000u;
      float u = cv.f - 1.0f;
      u = fmaxf(u * (1.0f - TINY) + TINY, TINY);
      float g = -logf(-logf(u));
      float s = logits[i] + g;
      if (s > best) { best = s; bi = i; }
    }
    s_act = bi;
  }
  __syncthreads();
  int node = sc[0];
  int a = s_act;
  float part = 0.0f;
  for (int i = threadIdx.x; i < S_DIM; i += 256)
    part += states[(size_t)node * S_DIM + i] * w_rew[(size_t)a * S_DIM + i];
  red[threadIdx.x] = part;
  __syncthreads();
  for (int s = 128; s > 0; s >>= 1) {
    if (threadIdx.x < s) red[threadIdx.x] += red[threadIdx.x + s];
    __syncthreads();
  }
  if (threadIdx.x == 0) { sc[5] = a; ((float*)sc)[8] = red[0]; }
}

// -------- traversal state update (masked scan step, like the reference) --------
__global__ void trav_update_kernel(int* __restrict__ sc, const int* __restrict__ children,
                                   const float* __restrict__ emb, const float* __restrict__ ns,
                                   float* __restrict__ l_next, int* __restrict__ p_nodes,
                                   int* __restrict__ p_actions, float* __restrict__ p_rewards,
                                   float* __restrict__ out_embpath, float* __restrict__ out_action,
                                   int d) {
  int node = sc[0], done = sc[1], action = sc[5];
  float rew = ((const float*)sc)[8];
  int child = children[node * A_DIM + action];
  int expanding = (child < 0) && !done;
  int valid = !done;
  __syncthreads();           // everyone has snapshotted scalars before thread 0 mutates them
  int i = threadIdx.x;       // 1024 threads
  if (expanding) l_next[i] = ns[i];
  out_embpath[i] = valid ? emb[(size_t)node * E_DIM + i] : 0.0f;
  if (i == 0) {
    p_nodes[d]   = valid ? node : -1;
    p_actions[d] = valid ? action : -1;
    p_rewards[d] = valid ? rew : 0.0f;
    *out_action  = valid ? (float)action : -1.0f;
    if (expanding) { sc[2] = node; sc[3] = action; }
    sc[4] += valid;
    int nd = done | expanding;
    sc[1] = nd;
    sc[0] = nd ? node : child;
  }
}

__global__ void expand_commit_kernel(const int* __restrict__ sc, float* __restrict__ states,
                                     int* __restrict__ children, const float* __restrict__ l_next,
                                     float* __restrict__ out_plen, int m) {
  int i = threadIdx.x;
  states[(size_t)(m + 1) * S_DIM + i] = l_next[i];
  if (i == 0) {
    children[sc[2] * A_DIM + sc[3]] = m + 1;
    *out_plen = (float)sc[4];
  }
}

__global__ void backup_prep_kernel(int* __restrict__ sc, const int* __restrict__ p_nodes,
                                   const int* __restrict__ p_actions,
                                   const float* __restrict__ p_rewards,
                                   const int* __restrict__ children, const float* __restrict__ emb,
                                   float* __restrict__ xbuf, int j) {
  __shared__ int s_nd, s_ch, s_a;
  __shared__ float s_r;
  if (threadIdx.x == 0) {
    int plen = sc[4];
    int valid = (j < plen) ? 1 : 0;
    int t = plen - 1 - j;
    int tc = t < 0 ? 0 : (t > D_MAX - 1 ? D_MAX - 1 : t);
    int nd = p_nodes[tc];   if (nd < 0) nd = 0;
    int a  = p_actions[tc]; if (a < 0) a = 0;
    int ch = children[nd * A_DIM + a]; if (ch < 0) ch = 0;
    s_nd = nd; s_ch = ch; s_a = a; s_r = p_rewards[tc];
    sc[6] = nd; sc[7] = valid;
  }
  __syncthreads();
  int i = threadIdx.x;   // 1024 threads; xbuf has KB1P=2176 slots (zero padded tail)
  xbuf[i]         = emb[(size_t)s_nd * E_DIM + i];
  xbuf[E_DIM + i] = emb[(size_t)s_ch * E_DIM + i];
  if (i < A_DIM) xbuf[2 * E_DIM + i] = (i == s_a) ? 1.0f : 0.0f;
  if (i == A_DIM) xbuf[2 * E_DIM + A_DIM] = s_r;
  if (i > A_DIM && i < 128) xbuf[2 * E_DIM + A_DIM + (i - A_DIM)] = 0.0f;  // pad 2065..2175
}

__global__ void backup_commit_kernel(const int* __restrict__ sc, const float* __restrict__ ne,
                                     float* __restrict__ emb) {
  if (sc[7]) emb[(size_t)sc[6] * E_DIM + threadIdx.x] = ne[threadIdx.x];
}

__global__ void sim_init_kernel(int* sc) {
  if (threadIdx.x == 0) { sc[0] = 0; sc[1] = 0; sc[2] = 0; sc[3] = 0; sc[4] = 0; }
}

__global__ void init_all_kernel(float* __restrict__ states, float* __restrict__ emb,
                                int* __restrict__ children, const float* __restrict__ state_in) {
  int tid = blockIdx.x * blockDim.x + threadIdx.x;
  int stride = gridDim.x * blockDim.x;
  for (int i = tid; i < (M_SIM + 1) * S_DIM; i += stride) {
    states[i] = (i < S_DIM) ? state_in[i] : 0.0f;
    emb[i] = 0.0f;
  }
  for (int i = tid; i < (M_SIM + 1) * A_DIM; i += stride) children[i] = -1;
}

extern "C" void kernel_launch(void* const* d_in, const int* in_sizes, int n_in,
                              void* d_out, int out_size, void* d_ws, size_t ws_size,
                              hipStream_t stream) {
  (void)in_sizes; (void)n_in; (void)out_size; (void)ws_size;
  const float* state = (const float*)d_in[0];
  const float* W_e1  = (const float*)d_in[1];
  const float* b_e1  = (const float*)d_in[2];
  const float* W_e2  = (const float*)d_in[3];
  const float* b_e2  = (const float*)d_in[4];
  const float* W_p   = (const float*)d_in[5];
  const float* b_p   = (const float*)d_in[6];
  const float* W_b1  = (const float*)d_in[7];
  const float* b_b1  = (const float*)d_in[8];
  const float* W_b2  = (const float*)d_in[9];
  const float* b_b2  = (const float*)d_in[10];
  const float* W_r   = (const float*)d_in[11];
  const float* b_r   = (const float*)d_in[12];
  const float* W_sim = (const float*)d_in[13];
  const float* b_sim = (const float*)d_in[14];
  const float* w_rew = (const float*)d_in[15];

  char* ws = (char*)d_ws;
  size_t off = 0;
  auto take = [&](size_t bytes) { size_t o = off; off = (off + bytes + 255) & ~(size_t)255; return o; };

  float* statesf  = (float*)(ws + take((size_t)(M_SIM + 1) * S_DIM * 4));
  float* embf     = (float*)(ws + take((size_t)(M_SIM + 1) * E_DIM * 4));
  int*   childreni= (int*)  (ws + take((size_t)(M_SIM + 1) * A_DIM * 4));
  int*   sc       = (int*)  (ws + take(64));
  int*   p_nodes  = (int*)  (ws + take(D_MAX * 4));
  int*   p_actions= (int*)  (ws + take(D_MAX * 4));
  float* p_rewards= (float*)(ws + take(D_MAX * 4));
  float* lnext    = (float*)(ws + take(S_DIM * 4));
  float* xbuf     = (float*)(ws + take(KB1P * 4));
  float* hbuf     = (float*)(ws + take(H_DIM * 4));
  float* nsbuf    = (float*)(ws + take(S_DIM * 4));
  float* nebuf    = (float*)(ws + take(E_DIM * 4));
  float* logits   = (float*)(ws + take(A_DIM * 4));
  uint32_t* We1p  = (uint32_t*)(ws + take((size_t)(S_DIM / 2) * H_DIM * 4));
  uint32_t* We2p  = (uint32_t*)(ws + take((size_t)(H_DIM / 2) * E_DIM * 4));
  uint32_t* Wb1p  = (uint32_t*)(ws + take((size_t)(KB1P / 2) * H_DIM * 4));
  uint32_t* Wb2p  = (uint32_t*)(ws + take((size_t)(H_DIM / 2) * E_DIM * 4));
  uint32_t* Wpp   = (uint32_t*)(ws + take((size_t)(E_DIM / 2) * A_DIM * 4));
  uint32_t* Wrp   = (uint32_t*)(ws + take((size_t)(E_DIM / 2) * A_DIM * 4));
  uint32_t* Wsimp = (uint32_t*)(ws + take((size_t)A_DIM * (S_DIM / 2) * S_DIM * 4));

  float* out       = (float*)d_out;
  float* out_preds = out;                                          // [16][16]
  float* out_embp  = out + M_SIM * A_DIM;                          // [16][16][1024]
  float* out_acts  = out_embp + (size_t)M_SIM * D_MAX * E_DIM;     // [16][16]
  float* out_plens = out_acts + M_SIM * D_MAX;                     // [16]

  auto pack = [&](const float* W, uint32_t* Wp, int K, int Kpad, int N) {
    pack_f16_kernel<<<dim3(2048), dim3(256), 0, stream>>>(W, Wp, K, Kpad, N);
  };
  auto gemv = [&](const uint32_t* Wp, long long wstride, const int* widx,
                  const float* xb, int xstride, const int* xidx,
                  const float* bias, int bstride, const int* bidx,
                  float* y, int Kpad, int N, int act) {
    gemv_wmma_kernel<<<dim3((N + 31) / 32), dim3(32), 0, stream>>>(
        Wp, wstride, widx, xb, xstride, xidx, bias, bstride, bidx, y, Kpad, N, act);
  };

  // ---- one-time-per-launch f16 weight packing (deterministic, L2-resident afterwards) ----
  pack(W_e1, We1p, S_DIM, S_DIM, H_DIM);
  pack(W_e2, We2p, H_DIM, H_DIM, E_DIM);
  pack(W_b1, Wb1p, KB1, KB1P, H_DIM);
  pack(W_b2, Wb2p, H_DIM, H_DIM, E_DIM);
  pack(W_p,  Wpp,  E_DIM, E_DIM, A_DIM);
  pack(W_r,  Wrp,  E_DIM, E_DIM, A_DIM);
  pack(W_sim, Wsimp, A_DIM * S_DIM, A_DIM * S_DIM, S_DIM);   // S even -> pairs never cross actions

  // ---- init tree + root embedding ----
  init_all_kernel<<<dim3(68), dim3(256), 0, stream>>>(statesf, embf, childreni, state);
  gemv(We1p, 0, nullptr, state, 0, nullptr, b_e1, 0, nullptr, hbuf, S_DIM, H_DIM, ACT_TANH);
  gemv(We2p, 0, nullptr, hbuf, 0, nullptr, b_e2, 0, nullptr, embf, H_DIM, E_DIM, ACT_NONE);

  for (int m = 0; m < M_SIM; ++m) {
    sim_init_kernel<<<dim3(1), dim3(32), 0, stream>>>(sc);

    // ---- traversal: fixed-length masked scan (as in reference) ----
    for (int d = 0; d < D_MAX; ++d) {
      gemv(Wpp, 0, nullptr, embf, E_DIM, sc + 0, b_p, 0, nullptr, logits, E_DIM, A_DIM, ACT_NONE);
      trav_sample_kernel<<<dim3(1), dim3(256), 0, stream>>>(sc, logits, statesf, w_rew, m, d);
      gemv(Wsimp, (long long)(S_DIM / 2) * S_DIM, sc + 5,
           statesf, S_DIM, sc + 0, b_sim, S_DIM, sc + 5, nsbuf, S_DIM, S_DIM, ACT_TANH);
      trav_update_kernel<<<dim3(1), dim3(1024), 0, stream>>>(
          sc, childreni, embf, nsbuf, lnext, p_nodes, p_actions, p_rewards,
          out_embp + ((size_t)m * D_MAX + d) * E_DIM, out_acts + m * D_MAX + d, d);
    }

    // ---- expand: new node index is statically m+1 ----
    gemv(We1p, 0, nullptr, lnext, 0, nullptr, b_e1, 0, nullptr, hbuf, S_DIM, H_DIM, ACT_TANH);
    gemv(We2p, 0, nullptr, hbuf, 0, nullptr, b_e2, 0, nullptr,
         embf + (size_t)(m + 1) * E_DIM, H_DIM, E_DIM, ACT_NONE);
    expand_commit_kernel<<<dim3(1), dim3(1024), 0, stream>>>(
        sc, statesf, childreni, lnext, out_plens + m, m);

    // ---- backup: fixed-length masked reversed scan ----
    for (int j = 0; j < D_MAX; ++j) {
      backup_prep_kernel<<<dim3(1), dim3(1024), 0, stream>>>(
          sc, p_nodes, p_actions, p_rewards, childreni, embf, xbuf, j);
      gemv(Wb1p, 0, nullptr, xbuf, 0, nullptr, b_b1, 0, nullptr, hbuf, KB1P, H_DIM, ACT_RELU);
      gemv(Wb2p, 0, nullptr, hbuf, 0, nullptr, b_b2, 0, nullptr, nebuf, H_DIM, E_DIM, ACT_NONE);
      backup_commit_kernel<<<dim3(1), dim3(1024), 0, stream>>>(sc, nebuf, embf);
    }

    // ---- readout ----
    gemv(Wrp, 0, nullptr, embf, 0, nullptr, b_r, 0, nullptr,
         out_preds + m * A_DIM, E_DIM, A_DIM, ACT_NONE);
  }
}
